// GatedESN_15539191677235
// MI455X (gfx1250) — matrix operand order repaired and checked by reading, not compiled, not run on hardware
//
#include <hip/hip_runtime.h>
#include <hip/hip_bf16.h>

// ---------------------------------------------------------------------------
// Gated ESN for MI455X (gfx1250), wave32 + WMMA bf16 + TDM staging.
//   B=32, T=2048, I=64, R=1024, H=256, leaky a=0.9
// Pipeline:
//   prep:  zero sync counters; convert u/weights to bf16 (transposed where the
//          GEMM needs W^T as the K-major B operand)
//   K1:    persistent ESN scan, 32 WGs x 128 thr, W^T slice staged to LDS by
//          the Tensor Data Mover (padded rows), per-step grid barrier,
//          state f32 in registers, dual WMMA accumulator chains
//   K2:    pca = res @ w_pca            (WMMA streaming GEMM, bf16 out)
//   K3:    pz  = pca @ wzp^T + bz       (WMMA streaming GEMM, f32 out)
//   K4:    persistent gate scan, 8 WGs x 128 thr, wzh^T in LDS, writes d_out
// ---------------------------------------------------------------------------

typedef __attribute__((ext_vector_type(16))) __bf16 v16bf;
typedef __attribute__((ext_vector_type(8)))  float  v8f;
typedef __attribute__((ext_vector_type(4)))  unsigned uint32x4;
typedef __attribute__((ext_vector_type(8)))  int      int32x8;
typedef __attribute__((ext_vector_type(4)))  int      int32x4;

union BPack { v16bf v; uint4 q[2]; };

constexpr int NB = 32;    // batch
constexpr int NT = 2048;  // time steps
constexpr int NI = 64;    // input dim
constexpr int NR = 1024;  // reservoir dim
constexpr int NH = 256;   // hidden dim
constexpr int WT_STRIDE = 40;  // padded LDS row stride (bf16) for 32-col slices

#if defined(__HIP_DEVICE_COMPILE__) && __has_builtin(__builtin_amdgcn_tensor_load_to_lds)
#define HAVE_TDM 1
#else
#define HAVE_TDM 0
#endif

__device__ __forceinline__ v8f wmma_bf16(v16bf a, v16bf b, v8f c) {
  return __builtin_amdgcn_wmma_f32_16x16x32_bf16(false, a, false, b, (short)0, c,
                                                 false, false);
}

__device__ __forceinline__ void grid_barrier(unsigned* ctr, unsigned target) {
  __threadfence();
  __syncthreads();
  if (threadIdx.x == 0) {
    atomicAdd(ctr, 1u);
    while (atomicAdd(ctr, 0u) < target) __builtin_amdgcn_s_sleep(2);
  }
  __syncthreads();
  __threadfence();
}

// ---------------------------------------------------------------------------
// prep kernels
// ---------------------------------------------------------------------------
__global__ void zero_u32_kernel(unsigned* p, int n) {
  int i = blockIdx.x * blockDim.x + threadIdx.x;
  if (i < n) p[i] = 0u;
}

__global__ void convert_bf16_kernel(const float* __restrict__ src,
                                    __bf16* __restrict__ dst, int n) {
  for (int i = blockIdx.x * blockDim.x + threadIdx.x; i < n;
       i += gridDim.x * blockDim.x)
    dst[i] = (__bf16)src[i];
}

// dst[c][r] = src[r][c]   (build K-major W^T operands)
__global__ void transpose_bf16_kernel(const float* __restrict__ src,
                                      __bf16* __restrict__ dst, int rows, int cols) {
  int n = rows * cols;
  for (int i = blockIdx.x * blockDim.x + threadIdx.x; i < n;
       i += gridDim.x * blockDim.x) {
    int r = i / cols, c = i - r * cols;
    dst[(size_t)c * rows + r] = (__bf16)src[i];
  }
}

// ---------------------------------------------------------------------------
// TDM: stage a [rows x 32] bf16 column slice of a K-major [rows x ldsrc]
// matrix into LDS at byte offset lds_off, padded to 80B rows (WT_STRIDE).
// D# layout per CDNA5 ISA 8.3/8.4. Issued by one wave; caller waits TENSORcnt.
// ---------------------------------------------------------------------------
#if HAVE_TDM
__device__ __forceinline__ void tdm_stage_slice(const __bf16* gsrc, unsigned lds_off,
                                                unsigned rows, unsigned ldsrc) {
  unsigned long long ga = (unsigned long long)gsrc;
  uint32x4 g0;
  g0.x = 1u;                                   // count=1 (valid user descriptor)
  g0.y = lds_off;                              // lds_addr (bytes)
  g0.z = (unsigned)(ga & 0xFFFFFFFFu);         // global_addr[31:0]
  g0.w = (unsigned)((ga >> 32) & 0x01FFFFFFu)  // global_addr[56:32]
         | 0x80000000u;                        // type=2 ("image")
  int32x8 g1 = {};
  g1[0] = (int)((1u << 16)     // data_size = 2B
              | (1u << 20)     // pad_enable
              | (3u << 22)     // pad_interval: every 16 dwords (one 64B row)
              | (3u << 25));   // pad_amount: 4 dwords (16B) -> 80B LDS rows
  g1[1] = (int)(ldsrc << 16);            // tensor_dim0[15:0]
  g1[2] = (int)(rows << 16);             // tensor_dim0[31:16]=0 | tensor_dim1[15:0]
  g1[3] = (int)(32u << 16);              // tensor_dim1[31:16]=0 | tile_dim0=32
  g1[4] = (int)rows;                     // tile_dim1 | tile_dim2=0
  g1[5] = (int)ldsrc;                    // tensor_dim0_stride[31:0]
  g1[6] = 0;                             // stride hi | dim1_stride lo
  g1[7] = 0;
  int32x4 g2 = {}, g3 = {};
#if __clang_major__ >= 23
  int32x8 g4 = {};
  __builtin_amdgcn_tensor_load_to_lds(g0, g1, g2, g3, g4, 0);
#else
  __builtin_amdgcn_tensor_load_to_lds(g0, g1, g2, g3, 0);
#endif
}
#endif

// ---------------------------------------------------------------------------
// K1: persistent LiESN reservoir scan.
//   grid = 32 WGs (each owns a 32-column slice of R), 128 threads = 4 waves,
//   wave w -> output tile (mtile = w&1 (batch rows), ntile = w>>1 (cols)).
//   x_t = 0.1*x + 0.9*tanh(u_t W_in^T + x_{t-1} W^T + b);  res[t] <- x_t (bf16)
// ---------------------------------------------------------------------------
__global__ void esn_scan_kernel(const __bf16* __restrict__ ub,    // [B][T][I]
                                const __bf16* __restrict__ winT,  // [I][R]
                                const __bf16* __restrict__ wT,    // [R][R]  wT[k][n]=w[n][k]
                                const float*  __restrict__ wbias, // [R]
                                __bf16* __restrict__ res,         // [T][B][R]
                                unsigned* __restrict__ ctr) {
  extern __shared__ __align__(16) __bf16 smem[];
  __bf16* Wt = smem;                        // [NR][WT_STRIDE]
  __bf16* Wu = smem + NR * WT_STRIDE;       // [NI][WT_STRIDE]

  const int tid = threadIdx.x;
  const int wg = blockIdx.x;
  const int n_base = wg * 32;

#if HAVE_TDM
  if (tid < 32) {  // wave 0 drives the Tensor Data Mover for the big W^T slice
    tdm_stage_slice(wT + n_base, (unsigned)__builtin_amdgcn_groupstaticsize(),
                    NR, NR);
    __builtin_amdgcn_s_wait_tensorcnt(0);
  }
#else
  {
    const unsigned* s = (const unsigned*)wT;
    unsigned* d = (unsigned*)Wt;
    for (int idx = tid; idx < NR * 16; idx += blockDim.x) {
      int row = idx >> 4, dw = idx & 15;
      d[row * (WT_STRIDE / 2) + dw] = s[(size_t)row * (NR / 2) + (n_base >> 1) + dw];
    }
  }
#endif
  {  // W_in^T slice is tiny: manual staging
    const unsigned* su = (const unsigned*)winT;
    unsigned* du = (unsigned*)Wu;
    for (int idx = tid; idx < NI * 16; idx += blockDim.x) {
      int row = idx >> 4, dw = idx & 15;
      du[row * (WT_STRIDE / 2) + dw] = su[(size_t)row * (NR / 2) + (n_base >> 1) + dw];
    }
  }
  __syncthreads();

  const int wave = tid >> 5, lane = tid & 31;
  const int mA = lane & 15;      // A-operand row owned by this lane
  const int hf = lane >> 4;      // lane half selects K sub-range
  const int nn = lane & 15;      // C/D column owned by this lane
  const int mtile = wave & 1;
  const int ntile = wave >> 1;
  const int bb = mtile * 16 + mA;         // batch row for A loads
  const int ncol = n_base + ntile * 16 + nn;
  const float bias_n = wbias[ncol];

  float xreg[8];
#pragma unroll
  for (int p = 0; p < 8; ++p) xreg[p] = 0.f;

  const __bf16* uArow = ub + (size_t)bb * (NT * NI);

  for (int t = 0; t < NT; ++t) {
    if (t + 1 < NT) __builtin_prefetch(uArow + (size_t)(t + 1) * NI, 0, 0);
    v8f acc0 = {}, acc1 = {};  // dual chains -> 2x ILP on the XDL pipe
    // input projection term, K = NI = 64
    {
      const __bf16* ar = uArow + (size_t)t * NI;
#pragma unroll
      for (int kb = 0; kb < NI / 32; kb += 2) {
        BPack a0, b0, a1, b1;
        a0.q[0] = *(const uint4*)(ar + kb * 32 + 8 * hf);
        a0.q[1] = *(const uint4*)(ar + kb * 32 + 16 + 8 * hf);
        a1.q[0] = *(const uint4*)(ar + kb * 32 + 32 + 8 * hf);
        a1.q[1] = *(const uint4*)(ar + kb * 32 + 48 + 8 * hf);
        const __bf16* br0 = Wu + (size_t)(kb * 32) * WT_STRIDE + ntile * 16 + nn;
        const __bf16* br1 = br0 + 32 * WT_STRIDE;
#pragma unroll
        for (int s = 0; s < 16; ++s) {
          int kk = (s < 8) ? 8 * hf + s : 8 + 8 * hf + s;
          b0.v[s] = br0[kk * WT_STRIDE];
          b1.v[s] = br1[kk * WT_STRIDE];
        }
        acc0 = wmma_bf16(a0.v, b0.v, acc0);
        acc1 = wmma_bf16(a1.v, b1.v, acc1);
      }
    }
    // recurrent term, K = NR = 1024 (x_{-1} == 0 -> skip at t==0)
    if (t > 0) {
      const __bf16* xr = res + ((size_t)(t - 1) * NB + bb) * NR;
#pragma unroll 2
      for (int kb = 0; kb < NR / 32; kb += 2) {
        BPack a0, b0, a1, b1;
        a0.q[0] = *(const uint4*)(xr + kb * 32 + 8 * hf);
        a0.q[1] = *(const uint4*)(xr + kb * 32 + 16 + 8 * hf);
        a1.q[0] = *(const uint4*)(xr + kb * 32 + 32 + 8 * hf);
        a1.q[1] = *(const uint4*)(xr + kb * 32 + 48 + 8 * hf);
        const __bf16* br0 = Wt + (size_t)(kb * 32) * WT_STRIDE + ntile * 16 + nn;
        const __bf16* br1 = br0 + 32 * WT_STRIDE;
#pragma unroll
        for (int s = 0; s < 16; ++s) {
          int kk = (s < 8) ? 8 * hf + s : 8 + 8 * hf + s;
          b0.v[s] = br0[kk * WT_STRIDE];
          b1.v[s] = br1[kk * WT_STRIDE];
        }
        acc0 = wmma_bf16(a0.v, b0.v, acc0);
        acc1 = wmma_bf16(a1.v, b1.v, acc1);
      }
    }
    // leaky update on register-resident f32 state; publish bf16 slice
    __bf16* orow = res + (size_t)t * NB * NR;
#pragma unroll
    for (int p = 0; p < 8; ++p) {
      int bo = mtile * 16 + p + 8 * hf;   // C/D row -> batch index
      float xn = 0.1f * xreg[p] + 0.9f * tanhf(acc0[p] + acc1[p] + bias_n);
      xreg[p] = xn;
      orow[(size_t)bo * NR + ncol] = (__bf16)xn;
    }
    grid_barrier(ctr, (unsigned)gridDim.x * (unsigned)(t + 1));
  }
}

// ---------------------------------------------------------------------------
// K2/K3: streaming row-parallel GEMM  out[m][n] = A[m][:] * Bm[:][n] (+bias)
//   A bf16 [M][lda], Bm bf16 K-major [K][ldb].
//   Each wave owns one M-tile and sweeps 8 N-tiles (128 cols) with 8 register
//   accumulators -> A rows read once per K-block (16x less L2 traffic than a
//   column-partitioned layout); small B absorbs redundancy in WGP$/L2.
//   grid.x = M/64 (4 waves = 4 M-tiles per block), grid.y = N/128.
// ---------------------------------------------------------------------------
__global__ void gemm_rows_kernel(const __bf16* __restrict__ A, int lda,
                                 const __bf16* __restrict__ Bm, int ldb, int Kdim,
                                 float* __restrict__ outF, __bf16* __restrict__ outB,
                                 int ldo, const float* __restrict__ bias) {
  const int tid = threadIdx.x, wave = tid >> 5, lane = tid & 31;
  const int mt = blockIdx.x * 4 + wave;
  const int n00 = blockIdx.y * 128;
  const int mA = lane & 15, hf = lane >> 4, nn = lane & 15;

  const __bf16* ar = A + (size_t)(mt * 16 + mA) * lda;
  v8f acc[8];
#pragma unroll
  for (int nt = 0; nt < 8; ++nt) acc[nt] = {};

  for (int kb = 0; kb < (Kdim >> 5); ++kb) {
    BPack a;
    a.q[0] = *(const uint4*)(ar + kb * 32 + 8 * hf);
    a.q[1] = *(const uint4*)(ar + kb * 32 + 16 + 8 * hf);
    const __bf16* brow = Bm + (size_t)(kb * 32) * ldb + n00 + nn;
#pragma unroll
    for (int nt = 0; nt < 8; ++nt) {
      BPack b;
#pragma unroll
      for (int s = 0; s < 16; ++s) {
        int kk = (s < 8) ? 8 * hf + s : 8 + 8 * hf + s;
        b.v[s] = brow[(size_t)kk * ldb + nt * 16];
      }
      acc[nt] = wmma_bf16(a.v, b.v, acc[nt]);
    }
  }
#pragma unroll
  for (int nt = 0; nt < 8; ++nt) {
#pragma unroll
    for (int p = 0; p < 8; ++p) {
      int mr = mt * 16 + p + 8 * hf;
      int nc = n00 + nt * 16 + nn;
      float v = acc[nt][p];
      if (bias) v += bias[nc];
      if (outF) outF[(size_t)mr * ldo + nc] = v;
      else      outB[(size_t)mr * ldo + nc] = (__bf16)v;
    }
  }
}

// ---------------------------------------------------------------------------
// K4: persistent gated recurrence.  8 WGs own 32-col slices of H.
//   z = sigmoid(pz[t] + h_{t-1} wzh^T);  h = (1-z) h_{t-1} + z * pca[t]
//   h history kept time-indexed (hs[T][B][H] bf16) to avoid cache staleness.
// ---------------------------------------------------------------------------
__global__ void gate_scan_kernel(const float*  __restrict__ pz,   // [T*B][H] (incl. bz)
                                 const __bf16* __restrict__ pca,  // [T*B][H]
                                 const __bf16* __restrict__ wzhT, // [H][H]
                                 __bf16* __restrict__ hs,         // [T*B][H]
                                 float* __restrict__ out,         // [B][T][H]
                                 unsigned* __restrict__ ctr) {
  __shared__ __align__(16) __bf16 Wh[NH * WT_STRIDE];
  const int tid = threadIdx.x;
  const int wg = blockIdx.x;
  const int n_base = wg * 32;
  {
    const unsigned* s = (const unsigned*)wzhT;
    unsigned* d = (unsigned*)Wh;
    for (int idx = tid; idx < NH * 16; idx += blockDim.x) {
      int row = idx >> 4, dw = idx & 15;
      d[row * (WT_STRIDE / 2) + dw] = s[(size_t)row * (NH / 2) + (n_base >> 1) + dw];
    }
  }
  __syncthreads();

  const int wave = tid >> 5, lane = tid & 31;
  const int mA = lane & 15, hf = lane >> 4, nn = lane & 15;
  const int mtile = wave & 1, ntile = wave >> 1;
  const int bb = mtile * 16 + mA;
  const int ng = n_base + ntile * 16 + nn;

  float hreg[8];
#pragma unroll
  for (int p = 0; p < 8; ++p) hreg[p] = 0.f;

  for (int t = 0; t < NT; ++t) {
    if (t + 1 < NT) {
      __builtin_prefetch(pz + ((size_t)(t + 1) * NB) * NH + ng, 0, 0);
      __builtin_prefetch(pca + ((size_t)(t + 1) * NB) * NH + ng, 0, 0);
    }
    v8f acc0 = {}, acc1 = {};
    if (t > 0) {  // h_{-1} == 0
      const __bf16* ar = hs + ((size_t)(t - 1) * NB + bb) * NH;
#pragma unroll
      for (int kb = 0; kb < NH / 32; kb += 2) {
        BPack a0, b0, a1, b1;
        a0.q[0] = *(const uint4*)(ar + kb * 32 + 8 * hf);
        a0.q[1] = *(const uint4*)(ar + kb * 32 + 16 + 8 * hf);
        a1.q[0] = *(const uint4*)(ar + kb * 32 + 32 + 8 * hf);
        a1.q[1] = *(const uint4*)(ar + kb * 32 + 48 + 8 * hf);
        const __bf16* br0 = Wh + (size_t)(kb * 32) * WT_STRIDE + ntile * 16 + nn;
        const __bf16* br1 = br0 + 32 * WT_STRIDE;
#pragma unroll
        for (int s = 0; s < 16; ++s) {
          int kk = (s < 8) ? 8 * hf + s : 8 + 8 * hf + s;
          b0.v[s] = br0[kk * WT_STRIDE];
          b1.v[s] = br1[kk * WT_STRIDE];
        }
        acc0 = wmma_bf16(a0.v, b0.v, acc0);
        acc1 = wmma_bf16(a1.v, b1.v, acc1);
      }
    }
#pragma unroll
    for (int p = 0; p < 8; ++p) {
      int bo = mtile * 16 + p + 8 * hf;
      size_t row = (size_t)t * NB + bo;
      float z = 1.f / (1.f + __expf(-(pz[row * NH + ng] + acc0[p] + acc1[p])));
      float pt = (float)pca[row * NH + ng];
      float h = (1.f - z) * hreg[p] + z * pt;
      hreg[p] = h;
      hs[row * NH + ng] = (__bf16)h;
      out[(size_t)bo * (NT * NH) + (size_t)t * NH + ng] = h;  // [B][T][H]
    }
    grid_barrier(ctr, (unsigned)gridDim.x * (unsigned)(t + 1));
  }
}

// ---------------------------------------------------------------------------
extern "C" void kernel_launch(void* const* d_in, const int* in_sizes, int n_in,
                              void* d_out, int out_size, void* d_ws, size_t ws_size,
                              hipStream_t stream) {
  const float* u      = (const float*)d_in[0];  // [B][T][I]
  const float* w_in   = (const float*)d_in[1];  // [R][I]
  const float* w      = (const float*)d_in[2];  // [R][R]
  const float* w_bias = (const float*)d_in[3];  // [R]
  const float* w_pca  = (const float*)d_in[4];  // [R][H]
  const float* wzp    = (const float*)d_in[5];  // [H][H]
  const float* wzh    = (const float*)d_in[6];  // [H][H]
  const float* bz     = (const float*)d_in[7];  // [H]
  float* out = (float*)d_out;

  char* ws = (char*)d_ws;
  size_t off = 0;
  auto alloc = [&](size_t bytes) -> void* {
    off = (off + 255) & ~(size_t)255;
    void* p = ws + off;
    off += bytes;
    return p;
  };
  unsigned* ctr   = (unsigned*)alloc(64 * sizeof(unsigned));
  __bf16* u_bf    = (__bf16*)alloc((size_t)NB * NT * NI * 2);
  __bf16* winT    = (__bf16*)alloc((size_t)NI * NR * 2);
  __bf16* wT      = (__bf16*)alloc((size_t)NR * NR * 2);
  __bf16* wpca_bf = (__bf16*)alloc((size_t)NR * NH * 2);
  __bf16* wzpT    = (__bf16*)alloc((size_t)NH * NH * 2);
  __bf16* wzhT    = (__bf16*)alloc((size_t)NH * NH * 2);
  __bf16* res     = (__bf16*)alloc((size_t)NT * NB * NR * 2);   // 128 MB
  __bf16* pca     = (__bf16*)alloc((size_t)NT * NB * NH * 2);   //  32 MB
  float*  pz      = (float*) alloc((size_t)NT * NB * NH * 4);   //  64 MB
  __bf16* hs      = (__bf16*)alloc((size_t)NT * NB * NH * 2);   //  32 MB

  // ---- prep -------------------------------------------------------------
  zero_u32_kernel<<<1, 64, 0, stream>>>(ctr, 64);
  convert_bf16_kernel<<<2048, 256, 0, stream>>>(u, u_bf, NB * NT * NI);
  transpose_bf16_kernel<<<256, 256, 0, stream>>>(w_in, winT, NR, NI);   // -> [I][R]
  transpose_bf16_kernel<<<1024, 256, 0, stream>>>(w, wT, NR, NR);       // -> W^T
  convert_bf16_kernel<<<256, 256, 0, stream>>>(w_pca, wpca_bf, NR * NH);
  transpose_bf16_kernel<<<64, 256, 0, stream>>>(wzp, wzpT, NH, NH);
  transpose_bf16_kernel<<<64, 256, 0, stream>>>(wzh, wzhT, NH, NH);

  // ---- K1: reservoir scan (sequential critical path) --------------------
  size_t lds1 = (size_t)(NR + NI) * WT_STRIDE * sizeof(__bf16);  // ~87 KB
  esn_scan_kernel<<<NR / 32, 128, lds1, stream>>>(u_bf, winT, wT, w_bias, res, ctr);

  // ---- K2: pca = res @ w_pca        M=T*B=65536, K=1024, N=256 ----------
  gemm_rows_kernel<<<dim3((NT * NB) / 64, NH / 128), 128, 0, stream>>>(
      res, NR, wpca_bf, NH, NR, nullptr, pca, NH, nullptr);

  // ---- K3: pz = pca @ wzp^T + bz    M=65536, K=256, N=256 ---------------
  gemm_rows_kernel<<<dim3((NT * NB) / 64, NH / 128), 128, 0, stream>>>(
      pca, NH, wzpT, NH, NH, pz, nullptr, NH, bz);

  // ---- K4: gate scan -> d_out -------------------------------------------
  gate_scan_kernel<<<NH / 32, 128, 0, stream>>>(pz, pca, wzhT, hs, out, ctr + 32);
}
